// LinearAttention_B_18854906430158
// MI455X (gfx1250) — matrix operand order repaired and checked by reading, not compiled, not run on hardware
//
#include <hip/hip_runtime.h>
#include <hip/hip_bf16.h>

// ---------------------------------------------------------------------------
// Linear attention for MI455X (gfx1250, wave32, WMMA 16x16x32 f16 -> f32 acc)
// + Tensor Data Mover staging for the per-head context reduction.
// ---------------------------------------------------------------------------

typedef __attribute__((ext_vector_type(16))) _Float16 v16h;
typedef __attribute__((ext_vector_type(8)))  float    v8f;
typedef __attribute__((ext_vector_type(4)))  _Float16 v4h;
typedef __attribute__((ext_vector_type(4)))  unsigned int u32x4;
typedef __attribute__((ext_vector_type(4)))  int          i32x4;
typedef __attribute__((ext_vector_type(8)))  int          i32x8;

#define B_   8
#define N_   16384
#define C_   256
#define H_   8
#define D_   32
#define MTOT (B_ * N_)   // 131072 rows

union Frag16 { v16h v; float4 f4[2]; };

// A-fragment 16x32 f16 from row-major [M,K] tile (ISA 7.12.2):
// lanes 0-15: row = lane, K 0-7 (v0-3) and 16-23 (v4-7)
// lanes 16-31: row = lane-16, K 8-15 and 24-31
__device__ __forceinline__ v16h load_a_frag(const _Float16* base, int ld) {
  int lane = threadIdx.x & 31;
  const _Float16* p = base + (size_t)(lane & 15) * ld + ((lane >> 4) << 3);
  Frag16 f;
  f.f4[0] = *(const float4*)(p);
  f.f4[1] = *(const float4*)(p + 16);
  return f.v;
}

// B-fragment 32x16 f16, loaded from row-major B^T[N,K] (i.e. weight W[o,c]):
// lanes 0-15: column n = lane, K 0-7 (v0-3) and 16-23 (v4-7)
// lanes 16-31: column n = lane-16, K 8-15 and 24-31
__device__ __forceinline__ v16h load_b_frag(const _Float16* base, int ld) {
  int lane = threadIdx.x & 31;
  const _Float16* p = base + (size_t)(lane & 15) * ld + ((lane >> 4) << 3);
  Frag16 f;
  f.f4[0] = *(const float4*)(p);
  f.f4[1] = *(const float4*)(p + 16);
  return f.v;
}

__device__ __forceinline__ v8f wmma_f16(v16h a, v16h b, v8f c) {
  return __builtin_amdgcn_wmma_f32_16x16x32_f16(
      /*neg_a=*/false, a, /*neg_b=*/false, b,
      /*c_mod=*/(short)0, c, /*reuse_a=*/false, /*reuse_b=*/false);
}

__device__ __forceinline__ float elu1(float x) {
  return x > 0.f ? x + 1.f : __expf(x);   // elu(x)+1
}

// ---------------------------------------------------------------------------
// TDM: async 2D tile load (64 rows x 32 cols of f16, row stride 256 elems)
// from global into LDS. Descriptor per CDNA5 ISA sec. 8 (D# groups 0/1).
// ---------------------------------------------------------------------------
__device__ __forceinline__ void tdm_load_tile_64x32(unsigned lds_off,
                                                    const _Float16* gptr) {
  unsigned long long ga = (unsigned long long)(uintptr_t)gptr;
  u32x4 g0;
  g0.x = 1u;                                   // count=1, user descriptor
  g0.y = lds_off;                              // lds_addr [63:32]
  g0.z = (unsigned)(ga & 0xffffffffu);         // global_addr [95:64]
  g0.w = (unsigned)((ga >> 32) & 0x1ffffffu)   // global_addr [120:96]
       | (2u << 30);                           // type=2 ("image") [127:126]
  i32x8 g1;
  g1[0] = 0x00010000;            // workgroup_mask=0, data_size=1 (2 bytes)
  g1[1] = (int)(256u << 16);     // tensor_dim0 = 256  (bits [63:48] = low16)
  g1[2] = (int)(16384u << 16);   // tensor_dim0 hi=0 | tensor_dim1 low16
  g1[3] = (int)(32u << 16);      // tensor_dim1 hi=0 | tile_dim0 = 32
  g1[4] = 64;                    // tile_dim1 = 64, tile_dim2 = 0
  g1[5] = 256;                   // tensor_dim0_stride = 256 (low 32)
  g1[6] = 0;                     // stride hi, tensor_dim1_stride low
  g1[7] = 0;
  i32x4 gz;
  gz[0] = 0; gz[1] = 0; gz[2] = 0; gz[3] = 0;
#if defined(__clang_major__) && (__clang_major__ >= 23)
  i32x8 gz8;
  gz8[0]=0; gz8[1]=0; gz8[2]=0; gz8[3]=0; gz8[4]=0; gz8[5]=0; gz8[6]=0; gz8[7]=0;
  __builtin_amdgcn_tensor_load_to_lds(g0, g1, gz, gz, gz8, 0);
#else
  __builtin_amdgcn_tensor_load_to_lds(g0, g1, gz, gz, 0);
#endif
}

// ---------------------------------------------------------------------------
// 0) fp32 -> fp16 conversion (x and weights), 4 elems/thread
// ---------------------------------------------------------------------------
__global__ __launch_bounds__(256) void cvt_f32_f16(const float* __restrict__ in,
                                                   _Float16* __restrict__ out,
                                                   int n) {
  int i = (blockIdx.x * 256 + threadIdx.x) * 4;
  if (i + 3 < n) {
    float4 v = *(const float4*)(in + i);
    v4h h = { (_Float16)v.x, (_Float16)v.y, (_Float16)v.z, (_Float16)v.w };
    *(v4h*)(out + i) = h;
  }
}

// ---------------------------------------------------------------------------
// 1) fused qkv projection: P = xh @ [Wq;Wkv]^T, epilogue applies elu+1 to q,k
//    tile: 64 rows x 64 cols per block, 4 waves; software-pipelined K loop
// ---------------------------------------------------------------------------
__global__ __launch_bounds__(128) void qkv_gemm(
    const _Float16* __restrict__ xh,    // [MTOT, 256]
    const _Float16* __restrict__ wall,  // [768, 256]  rows: Wq | Wkv
    _Float16* __restrict__ qbuf,        // [MTOT, 256]  elu+1 applied
    _Float16* __restrict__ kbuf,        // [MTOT, 256]  elu+1 applied
    _Float16* __restrict__ vbuf)        // [MTOT, 256]
{
  int wave = threadIdx.x >> 5;
  int lane = threadIdx.x & 31;
  size_t row0 = (size_t)blockIdx.x * 64 + wave * 16;
  int col0 = blockIdx.y * 64;

  const _Float16* abase = xh + row0 * C_;
  const _Float16* wb0 = wall + (size_t)(col0 +  0) * C_;
  const _Float16* wb1 = wall + (size_t)(col0 + 16) * C_;
  const _Float16* wb2 = wall + (size_t)(col0 + 32) * C_;
  const _Float16* wb3 = wall + (size_t)(col0 + 48) * C_;

  v8f acc[4] = {};
  // prologue: stage K-step 0 fragments
  v16h a  = load_a_frag(abase, C_);
  v16h b0 = load_b_frag(wb0, C_);
  v16h b1 = load_b_frag(wb1, C_);
  v16h b2 = load_b_frag(wb2, C_);
  v16h b3 = load_b_frag(wb3, C_);
#pragma unroll
  for (int kk = 0; kk < C_; kk += 32) {
    v16h an = a, n0 = b0, n1 = b1, n2 = b2, n3 = b3;
    if (kk + 32 < C_) {           // stage next K-step before consuming current
      __builtin_prefetch(abase + kk + 64, 0, 3);
      an = load_a_frag(abase + kk + 32, C_);
      n0 = load_b_frag(wb0 + kk + 32, C_);
      n1 = load_b_frag(wb1 + kk + 32, C_);
      n2 = load_b_frag(wb2 + kk + 32, C_);
      n3 = load_b_frag(wb3 + kk + 32, C_);
    }
    acc[0] = wmma_f16(a, b0, acc[0]);
    acc[1] = wmma_f16(a, b1, acc[1]);
    acc[2] = wmma_f16(a, b2, acc[2]);
    acc[3] = wmma_f16(a, b3, acc[3]);
    a = an; b0 = n0; b1 = n1; b2 = n2; b3 = n3;
  }

  int mofs = (lane >> 4) << 3;   // 0 or 8
  int nofs = lane & 15;
#pragma unroll
  for (int j = 0; j < 4; ++j) {
    int o = col0 + j * 16 + nofs;
#pragma unroll
    for (int r = 0; r < 8; ++r) {
      size_t n = row0 + r + mofs;
      float val = acc[j][r];
      if (o < 256)       qbuf[n * C_ + o]         = (_Float16)elu1(val);
      else if (o < 512)  kbuf[n * C_ + (o - 256)] = (_Float16)elu1(val);
      else               vbuf[n * C_ + (o - 512)] = (_Float16)val;
    }
  }
}

// ---------------------------------------------------------------------------
// 2) per-(b,h): ctxT[e][d] = sum_n v[n,e]*k[n,d]  (stored transposed so it
//    loads directly as a WMMA B-fragment), plus k_sum[d].
//    Tiles staged LDS via Tensor Data Mover (TENSOR_LOAD_TO_LDS), completion
//    via s_wait_tensorcnt + workgroup barrier. Output is only 32x32/head.
// ---------------------------------------------------------------------------
__global__ __launch_bounds__(1024) void ctx_kernel(
    const _Float16* __restrict__ kbuf,  // [MTOT, 256]
    const _Float16* __restrict__ vbuf,  // [MTOT, 256]
    _Float16* __restrict__ ctxh,        // [B*H][32][32]  (e-major, f16)
    float* __restrict__ ksum)           // [B*H][32]
{
  __shared__ _Float16 lk[64][32];
  __shared__ _Float16 lv[64][32];

  int bh = blockIdx.x;
  int b = bh >> 3, head = bh & 7;
  int tid = threadIdx.x;
  int e  = tid >> 5;    // v feature
  int dd = tid & 31;    // k feature

  unsigned lk_off = (unsigned)(uintptr_t)(void*)&lk[0][0];
  unsigned lv_off = (unsigned)(uintptr_t)(void*)&lv[0][0];

  float acc = 0.f;
  float ks  = 0.f;

  for (int n0 = 0; n0 < N_; n0 += 64) {
    if (tid < 32) {   // wave 0 issues the two tile DMAs for the workgroup
      size_t rowoff = ((size_t)b * N_ + n0) * C_ + head * D_;
      tdm_load_tile_64x32(lk_off, kbuf + rowoff);
      tdm_load_tile_64x32(lv_off, vbuf + rowoff);
      __builtin_amdgcn_s_wait_tensorcnt(0);
    }
    __syncthreads();
#pragma unroll 8
    for (int i = 0; i < 64; ++i) {
      acc += (float)lk[i][dd] * (float)lv[i][e];
      if (tid < 32) ks += (float)lk[i][tid];
    }
    __syncthreads();
  }

  ctxh[((size_t)bh * 32 + e) * 32 + dd] = (_Float16)acc;
  if (tid < 32) ksum[bh * 32 + tid] = ks;
}

// ---------------------------------------------------------------------------
// 3) attn = (q @ ctx) * Dinv,  K = d = 32 -> exactly one WMMA per 16x16 tile.
//    Dinv computed per row by lanes 0-15, broadcast via wave shuffle.
//    Stores merge heads back to [B,N,C] layout (c = head*32 + e).
// ---------------------------------------------------------------------------
__global__ __launch_bounds__(256) void attn_kernel(
    const _Float16* __restrict__ qbuf,  // [MTOT, 256] (elu+1 applied)
    const _Float16* __restrict__ ctxh,  // [B*H][32][32] e-major
    const float* __restrict__ ksum,     // [B*H][32]
    _Float16* __restrict__ obuf)        // [MTOT, 256]
{
  int bh = blockIdx.y;
  int b = bh >> 3, head = bh & 7;
  int wave = threadIdx.x >> 5;
  int lane = threadIdx.x & 31;
  size_t n0 = (size_t)blockIdx.x * 128 + wave * 16;   // 16 rows per wave

  const _Float16* qbase = qbuf + ((size_t)b * N_ + n0) * C_ + head * D_;

  // Dinv for row (lane&15)
  float dv;
  {
    int r = lane & 15;
    const _Float16* qr = qbase + (size_t)r * C_;
    const float* ksp = ksum + bh * D_;
    float s = 0.f;
#pragma unroll
    for (int i = 0; i < D_; ++i) s += (float)qr[i] * ksp[i];
    dv = 1.0f / fmaxf(s, 1e-6f);
  }

  v16h a = load_a_frag(qbase, C_);
  const _Float16* cb = ctxh + (size_t)bh * (D_ * D_);
  v16h b0 = load_b_frag(cb, D_);
  v16h b1 = load_b_frag(cb + 16 * D_, D_);
  v8f acc0 = {}, acc1 = {};
  acc0 = wmma_f16(a, b0, acc0);
  acc1 = wmma_f16(a, b1, acc1);

  int mofs = (lane >> 4) << 3;
  int nofs = lane & 15;
#pragma unroll
  for (int r = 0; r < 8; ++r) {
    int m = r + mofs;
    float d = __shfl(dv, m, 32);          // Dinv of row m (held by lane m)
    size_t n = (size_t)b * N_ + n0 + m;
    obuf[n * C_ + head * D_ + nofs]      = (_Float16)(acc0[r] * d);
    obuf[n * C_ + head * D_ + 16 + nofs] = (_Float16)(acc1[r] * d);
  }
}

// ---------------------------------------------------------------------------
// 4) output projection: out = obuf @ Wproj^T + bproj  (fp32 out)
// ---------------------------------------------------------------------------
__global__ __launch_bounds__(128) void proj_gemm(
    const _Float16* __restrict__ obuf,   // [MTOT, 256]
    const _Float16* __restrict__ wph,    // [256, 256]
    const float* __restrict__ bproj,     // [256]
    float* __restrict__ out)             // [MTOT, 256] fp32
{
  int wave = threadIdx.x >> 5;
  int lane = threadIdx.x & 31;
  size_t row0 = (size_t)blockIdx.x * 64 + wave * 16;
  int col0 = blockIdx.y * 64;

  const _Float16* abase = obuf + row0 * C_;
  const _Float16* wb0 = wph + (size_t)(col0 +  0) * C_;
  const _Float16* wb1 = wph + (size_t)(col0 + 16) * C_;
  const _Float16* wb2 = wph + (size_t)(col0 + 32) * C_;
  const _Float16* wb3 = wph + (size_t)(col0 + 48) * C_;

  v8f acc[4] = {};
  v16h a  = load_a_frag(abase, C_);
  v16h b0 = load_b_frag(wb0, C_);
  v16h b1 = load_b_frag(wb1, C_);
  v16h b2 = load_b_frag(wb2, C_);
  v16h b3 = load_b_frag(wb3, C_);
#pragma unroll
  for (int kk = 0; kk < C_; kk += 32) {
    v16h an = a, n0 = b0, n1 = b1, n2 = b2, n3 = b3;
    if (kk + 32 < C_) {
      __builtin_prefetch(abase + kk + 64, 0, 3);
      an = load_a_frag(abase + kk + 32, C_);
      n0 = load_b_frag(wb0 + kk + 32, C_);
      n1 = load_b_frag(wb1 + kk + 32, C_);
      n2 = load_b_frag(wb2 + kk + 32, C_);
      n3 = load_b_frag(wb3 + kk + 32, C_);
    }
    acc[0] = wmma_f16(a, b0, acc[0]);
    acc[1] = wmma_f16(a, b1, acc[1]);
    acc[2] = wmma_f16(a, b2, acc[2]);
    acc[3] = wmma_f16(a, b3, acc[3]);
    a = an; b0 = n0; b1 = n1; b2 = n2; b3 = n3;
  }

  int mofs = (lane >> 4) << 3;
  int nofs = lane & 15;
#pragma unroll
  for (int j = 0; j < 4; ++j) {
    int o = col0 + j * 16 + nofs;
    float bb = bproj[o];
#pragma unroll
    for (int r = 0; r < 8; ++r) {
      size_t n = row0 + r + mofs;
      out[n * C_ + o] = acc[j][r] + bb;
    }
  }
}

// ---------------------------------------------------------------------------
extern "C" void kernel_launch(void* const* d_in, const int* in_sizes, int n_in,
                              void* d_out, int out_size, void* d_ws, size_t ws_size,
                              hipStream_t stream) {
  const float* x     = (const float*)d_in[0];   // [B,N,C]
  const float* Wq    = (const float*)d_in[1];   // [256,256]
  const float* Wkv   = (const float*)d_in[2];   // [512,256]
  const float* Wproj = (const float*)d_in[3];   // [256,256]
  const float* bproj = (const float*)d_in[4];   // [256]
  float* out = (float*)d_out;

  // workspace carve-up (all offsets 256B aligned)
  char* w = (char*)d_ws;
  size_t off = 0;
  auto carve = [&](size_t bytes) {
    void* p = w + off;
    off += (bytes + 255) & ~(size_t)255;
    return p;
  };
  const size_t big = (size_t)MTOT * C_ * sizeof(_Float16);   // 67.1 MB
  _Float16* xh   = (_Float16*)carve(big);
  _Float16* wall = (_Float16*)carve((size_t)768 * C_ * sizeof(_Float16));
  _Float16* wph  = (_Float16*)carve((size_t)C_ * C_ * sizeof(_Float16));
  _Float16* qb   = (_Float16*)carve(big);
  _Float16* kb   = (_Float16*)carve(big);
  _Float16* vb   = (_Float16*)carve(big);
  _Float16* ctxh = (_Float16*)carve((size_t)B_ * H_ * D_ * D_ * sizeof(_Float16));
  float*    ksum = (float*)carve((size_t)B_ * H_ * D_ * sizeof(float));
  _Float16* ob   = xh;   // xh dead after qkv_gemm; reuse for attn output

  // 0) conversions
  {
    int n = MTOT * C_;
    cvt_f32_f16<<<dim3(n / 1024), dim3(256), 0, stream>>>(x, xh, n);
    cvt_f32_f16<<<dim3((256 * 256) / 1024), dim3(256), 0, stream>>>(Wq, wall, 256 * 256);
    cvt_f32_f16<<<dim3((512 * 256) / 1024), dim3(256), 0, stream>>>(Wkv, wall + 256 * 256, 512 * 256);
    cvt_f32_f16<<<dim3((256 * 256) / 1024), dim3(256), 0, stream>>>(Wproj, wph, 256 * 256);
  }
  // 1) fused qkv projection + feature map
  qkv_gemm<<<dim3(MTOT / 64, 768 / 64), dim3(128), 0, stream>>>(xh, wall, qb, kb, vb);
  // 2) per-head context (transposed) + k_sum, TDM-staged
  ctx_kernel<<<dim3(B_ * H_), dim3(1024), 0, stream>>>(kb, vb, ctxh, ksum);
  // 3) attention + normalizer, merge heads
  attn_kernel<<<dim3(N_ / 128, B_ * H_), dim3(256), 0, stream>>>(qb, ctxh, ksum, ob);
  // 4) output projection + bias
  proj_gemm<<<dim3(MTOT / 64, C_ / 64), dim3(128), 0, stream>>>(ob, wph, bproj, out);
}
